// Net_48318382080266
// MI455X (gfx1250) — compile-verified, matrix operand order checked by reading, not run on hardware
//
#include <hip/hip_runtime.h>
#include <hip/hip_bf16.h>
#include <math.h>

// Problem constants (match reference)
#define NN    50000
#define DD    80
#define EE    800000
#define EDD   16
#define PED   37
#define GG    256
#define MSGD  176      // 2*DD + EDD
#define AGGD  704      // 4*MSGD
#define SCLD  2112     // 3*AGGD
#define NLAYERS 4
#define AVG_LOG_DEG 1.40592128174808f
#define EPS_BN 1e-5f

typedef __attribute__((ext_vector_type(16))) _Float16 v16h;
typedef __attribute__((ext_vector_type(8)))  float    v8f;

// ---- ordered-uint encoding so unsigned atomicMax/Min == float max/min ----
__device__ __forceinline__ unsigned floatFlip(float f) {
  unsigned u = __float_as_uint(f);
  return u ^ ((u & 0x80000000u) ? 0xFFFFFFFFu : 0x80000000u);
}
__device__ __forceinline__ float floatUnflip(unsigned u) {
  return __uint_as_float(u ^ ((u & 0x80000000u) ? 0x80000000u : 0xFFFFFFFFu));
}

// ---------------------------------------------------------------- init ----
__global__ void init_minmax_kernel(unsigned* __restrict__ maxb,
                                   unsigned* __restrict__ minb, int total) {
  int t = blockIdx.x * blockDim.x + threadIdx.x;
  if (t >= total) return;
  maxb[t] = 0x00800000u;  // floatFlip(-FLT_MAX)
  minb[t] = 0xFF7FFFFFu;  // floatFlip(+FLT_MAX)
}

// ------------------------------------------------- xp = x + PE@pe_w + b ----
__global__ void pe_embed_kernel(const float* __restrict__ x,
                                const float* __restrict__ PE,
                                const float* __restrict__ pw,
                                const float* __restrict__ pb,
                                float* __restrict__ xp) {
  int t = blockIdx.x * blockDim.x + threadIdx.x;
  if (t >= NN * DD) return;
  int n = t / DD, f = t % DD;
  const float* per = PE + n * PED;
  float acc = pb[f];
#pragma unroll
  for (int k = 0; k < PED; ++k) acc += per[k] * pw[k * DD + f];
  xp[t] = x[t] + acc;
}

// ----------------------------------------- scatter edge messages to tgt ----
__global__ void edge_agg_kernel(const int* __restrict__ ei,
                                const float* __restrict__ eattr,
                                const float* __restrict__ xp,
                                float* __restrict__ sumb,
                                float* __restrict__ sqb,
                                unsigned* __restrict__ maxb,
                                unsigned* __restrict__ minb,
                                float* __restrict__ cntb) {
  int t = blockIdx.x * blockDim.x + threadIdx.x;
  if (t >= EE * MSGD) return;
  int e = t / MSGD, f = t % MSGD;
  int sn = ei[e];           // src
  int tn = ei[EE + e];      // tgt
  float v;
  if (f < DD)            v = xp[tn * DD + f];            // x_i
  else if (f < 2 * DD)   v = xp[sn * DD + (f - DD)];     // x_j
  else                   v = eattr[e * EDD + (f - 2 * DD)];
  int o = tn * MSGD + f;
  atomicAdd(&sumb[o], v);
  atomicAdd(&sqb[o], v * v);
  unsigned fv = floatFlip(v);
  atomicMax(&maxb[o], fv);
  atomicMin(&minb[o], fv);
  if (f == 0) atomicAdd(&cntb[tn], 1.0f);
}

// ----------------------- agg = [mean,min,max,std] (f16) + degree scalers ----
__global__ void finalize_agg_kernel(const float* __restrict__ sumb,
                                    const float* __restrict__ sqb,
                                    const unsigned* __restrict__ maxb,
                                    const unsigned* __restrict__ minb,
                                    const float* __restrict__ cntb,
                                    _Float16* __restrict__ aggh,
                                    float* __restrict__ s1,
                                    float* __restrict__ s2) {
  int t = blockIdx.x * blockDim.x + threadIdx.x;
  if (t >= NN * MSGD) return;
  int n = t / MSGD, f = t % MSGD;
  float c = cntb[n];
  float d = fmaxf(c, 1.0f);
  float mean = sumb[t] / d;
  float var  = sqb[t] / d - mean * mean;
  var = var > 0.0f ? var : 0.0f;
  float stdv = sqrtf(var + 1e-5f);
  float mx = (c > 0.0f) ? floatUnflip(maxb[t]) : 0.0f;
  float mn = (c > 0.0f) ? floatUnflip(minb[t]) : 0.0f;
  size_t base = (size_t)n * AGGD;
  aggh[base + f]            = (_Float16)mean;
  aggh[base + MSGD + f]     = (_Float16)mn;
  aggh[base + 2 * MSGD + f] = (_Float16)mx;
  aggh[base + 3 * MSGD + f] = (_Float16)stdv;
  if (f == 0) {
    float logd = logf(d + 1.0f);
    s1[n] = logd / AVG_LOG_DEG;
    s2[n] = AVG_LOG_DEG / logd;
  }
}

// ------------------- conv_w [2112,80] f32 -> 3x transposed [80,704] f16 ----
__global__ void prep_w_kernel(const float* __restrict__ w,
                              _Float16* __restrict__ wt) {
  int t = blockIdx.x * blockDim.x + threadIdx.x;
  if (t >= SCLD * DD) return;
  int k = t / DD, n = t % DD;
  int part = k / AGGD, kk = k % AGGD;
  wt[((size_t)part * DD + n) * AGGD + kk] = (_Float16)w[t];
}

// ------------------------------- WMMA GEMM: out = agg@W1 + s1*agg@W2 + ... ----
// One wave per 16x16 tile; 5 waves/block cover the 80 output columns for one
// 16-row slab. K = 704 = 11 unrolled iterations of 2x32 -> 66 WMMAs / wave.
__global__ __launch_bounds__(160) void pna_gemm_wmma(
    const _Float16* __restrict__ aggh,   // [NN, AGGD] row-major f16
    const _Float16* __restrict__ wt,     // [3][DD][AGGD] f16 (W^T per part)
    const float* __restrict__ s1,
    const float* __restrict__ s2,
    const float* __restrict__ bias,      // [DD]
    float* __restrict__ out)             // [NN, DD] f32
{
  const int lane = threadIdx.x & 31;
  const int wave = threadIdx.x >> 5;     // 0..4 -> column tile
  const int m0 = blockIdx.x * 16;
  const int n0 = wave * 16;
  const int r  = lane & 15;
  const int hi = lane >> 4;
  const int kh = hi ? 8 : 0;

  const _Float16* __restrict__ arow = aggh + (size_t)(m0 + r) * AGGD;
  const _Float16* __restrict__ b1r  = wt + (size_t)(0 * DD + n0 + r) * AGGD;
  const _Float16* __restrict__ b2r  = wt + (size_t)(1 * DD + n0 + r) * AGGD;
  const _Float16* __restrict__ b3r  = wt + (size_t)(2 * DD + n0 + r) * AGGD;

  v8f acc1 = {}, acc2 = {}, acc3 = {};

  // Unroll x2: two K-chunks (6 WMMAs, 16 clause-grouped b128 loads) per
  // iteration so loads of chunk k+1 overlap the XDL occupancy of chunk k.
#pragma unroll 2
  for (int k0 = 0; k0 < AGGD; k0 += 32) {
    v16h a = {}, fb1 = {}, fb2 = {}, fb3 = {};
#pragma unroll
    for (int p = 0; p < 8; ++p) {
      // CDNA5 16-bit A layout: VGPR p holds K pair; hi half-lanes +8.
      const int kk = k0 + (p < 4 ? 2 * p : 16 + 2 * (p - 4)) + kh;
      a[2 * p]   = arow[kk]; a[2 * p + 1]   = arow[kk + 1];
      fb1[2 * p] = b1r[kk];  fb1[2 * p + 1] = b1r[kk + 1];
      fb2[2 * p] = b2r[kk];  fb2[2 * p + 1] = b2r[kk + 1];
      fb3[2 * p] = b3r[kk];  fb3[2 * p + 1] = b3r[kk + 1];
    }
    __builtin_prefetch(arow + k0 + 64, 0, 1);   // global_prefetch_b8, 128B ahead
    acc1 = __builtin_amdgcn_wmma_f32_16x16x32_f16(false, a, false, fb1,
                                                  (short)0, acc1, false, false);
    acc2 = __builtin_amdgcn_wmma_f32_16x16x32_f16(false, a, false, fb2,
                                                  (short)0, acc2, false, false);
    acc3 = __builtin_amdgcn_wmma_f32_16x16x32_f16(false, a, false, fb3,
                                                  (short)0, acc3, false, false);
  }

  const int col = n0 + r;
  const float bv = bias[col];
#pragma unroll
  for (int j = 0; j < 8; ++j) {                  // C/D row = j + 8*hi
    const int row = m0 + j + 8 * hi;
    const float v = acc1[j] + s1[row] * acc2[j] + s2[row] * acc3[j] + bv;
    out[(size_t)row * DD + col] = v;
  }
}

// ------------------------------------------------------- batch-norm ----
__global__ void bn_stats_kernel(const float* __restrict__ out,
                                float* __restrict__ stats, int total) {
  __shared__ float ls[2 * DD];
  for (int i = threadIdx.x; i < 2 * DD; i += blockDim.x) ls[i] = 0.0f;
  __syncthreads();
  int t = blockIdx.x * blockDim.x + threadIdx.x;
  if (t < total) {
    float v = out[t];
    int f = t % DD;
    atomicAdd(&ls[f], v);
    atomicAdd(&ls[DD + f], v * v);
  }
  __syncthreads();
  for (int i = threadIdx.x; i < 2 * DD; i += blockDim.x)
    atomicAdd(&stats[i], ls[i]);
}

__global__ void bn_finalize_kernel(float* __restrict__ stats,
                                   const float* __restrict__ g,
                                   const float* __restrict__ beta,
                                   float invN) {
  int f = threadIdx.x;
  if (f >= DD) return;
  float mu  = stats[f] * invN;
  float var = stats[DD + f] * invN - mu * mu;
  float inv = rsqrtf(var + EPS_BN) * g[f];
  stats[2 * DD + f] = inv;                // scale
  stats[3 * DD + f] = beta[f] - mu * inv; // shift
}

__global__ void bn_apply_kernel(const float* __restrict__ out,
                                const float* __restrict__ stats,
                                float* __restrict__ x, int total) {
  int t = blockIdx.x * blockDim.x + threadIdx.x;
  if (t >= total) return;
  int f = t % DD;
  float v = out[t] * stats[2 * DD + f] + stats[3 * DD + f];
  v = v > 0.0f ? v : 0.0f;
  x[t] = v + x[t];                        // ReLU + residual
}

// ------------------------------------------------------- pooling + head ----
__global__ void pool_kernel(const float* __restrict__ x,
                            const int* __restrict__ batch,
                            float* __restrict__ pooled,
                            float* __restrict__ gcnt, int total) {
  int t = blockIdx.x * blockDim.x + threadIdx.x;
  if (t >= total) return;
  int n = t / DD, f = t % DD;
  int g = batch[n];
  atomicAdd(&pooled[(size_t)g * DD + f], x[t]);
  if (f == 0) atomicAdd(&gcnt[g], 1.0f);
}

__global__ void head_kernel(const float* __restrict__ pooled,
                            const float* __restrict__ gcnt,
                            const float* __restrict__ w1, const float* __restrict__ b1,
                            const float* __restrict__ w2, const float* __restrict__ b2,
                            const float* __restrict__ w3, const float* __restrict__ b3,
                            float* __restrict__ outp) {
  int g = blockIdx.x * blockDim.x + threadIdx.x;
  if (g >= GG) return;
  float d = fmaxf(gcnt[g], 1.0f);
  float h1[40];
  for (int j = 0; j < 40; ++j) {
    float a = b1[j];
    for (int i = 0; i < DD; ++i) a += (pooled[(size_t)g * DD + i] / d) * w1[i * 40 + j];
    h1[j] = a > 0.0f ? a : 0.0f;
  }
  float h2[20];
  for (int j = 0; j < 20; ++j) {
    float a = b2[j];
    for (int i = 0; i < 40; ++i) a += h1[i] * w2[i * 20 + j];
    h2[j] = a > 0.0f ? a : 0.0f;
  }
  float a = b3[0];
  for (int i = 0; i < 20; ++i) a += h2[i] * w3[i];
  outp[g] = a;
}

// ==========================================================================
extern "C" void kernel_launch(void* const* d_in, const int* in_sizes, int n_in,
                              void* d_out, int out_size, void* d_ws, size_t ws_size,
                              hipStream_t stream) {
  const float* x_in   = (const float*)d_in[0];
  const int*   ei     = (const int*)d_in[1];
  const float* eattr  = (const float*)d_in[2];
  const float* PE     = (const float*)d_in[3];
  // d_in[4] = snorm (unused by reference)
  const int*   batch  = (const int*)d_in[5];
  const float* pe_w   = (const float*)d_in[6];
  const float* pe_b   = (const float*)d_in[7];
  const float* conv_w = (const float*)d_in[8];
  const float* conv_b = (const float*)d_in[9];
  const float* bn_g   = (const float*)d_in[10];
  const float* bn_b   = (const float*)d_in[11];
  const float* w1 = (const float*)d_in[12]; const float* b1 = (const float*)d_in[13];
  const float* w2 = (const float*)d_in[14]; const float* b2 = (const float*)d_in[15];
  const float* w3 = (const float*)d_in[16]; const float* b3 = (const float*)d_in[17];
  float* outp = (float*)d_out;

  // Workspace carving (~248 MB total; GEMM output aliases the dead sum buffer)
  char* p = (char*)d_ws;
  auto carve = [&](size_t bytes) -> char* {
    char* r = p; p += (bytes + 255) & ~(size_t)255; return r;
  };
  float*    x_cur = (float*)carve(sizeof(float) * (size_t)NN * DD);
  float*    xp    = (float*)carve(sizeof(float) * (size_t)NN * DD);
  float*    sumb  = (float*)carve(sizeof(float) * (size_t)NN * MSGD);
  float*    sqb   = (float*)carve(sizeof(float) * (size_t)NN * MSGD);
  unsigned* maxb  = (unsigned*)carve(sizeof(unsigned) * (size_t)NN * MSGD);
  unsigned* minb  = (unsigned*)carve(sizeof(unsigned) * (size_t)NN * MSGD);
  float*    cntb  = (float*)carve(sizeof(float) * NN);
  _Float16* aggh  = (_Float16*)carve(sizeof(_Float16) * (size_t)NN * AGGD);
  float*    s1    = (float*)carve(sizeof(float) * NN);
  float*    s2    = (float*)carve(sizeof(float) * NN);
  _Float16* wt    = (_Float16*)carve(sizeof(_Float16) * 3 * DD * AGGD);
  float*    bnst  = (float*)carve(sizeof(float) * 4 * DD);
  float*    pooled= (float*)carve(sizeof(float) * GG * DD);
  float*    gcnt  = (float*)carve(sizeof(float) * GG);
  float*    outbuf = sumb;  // reuse: sum dead after finalize, re-zeroed next layer

  const int BS = 256;
  auto nb = [](long long total, int bs) { return (unsigned)((total + bs - 1) / bs); };

  (void)hipMemcpyAsync(x_cur, x_in, sizeof(float) * (size_t)NN * DD,
                       hipMemcpyDeviceToDevice, stream);

  for (int l = 0; l < NLAYERS; ++l) {
    (void)hipMemsetAsync(sumb, 0, sizeof(float) * (size_t)NN * MSGD, stream);
    (void)hipMemsetAsync(sqb,  0, sizeof(float) * (size_t)NN * MSGD, stream);
    (void)hipMemsetAsync(cntb, 0, sizeof(float) * NN, stream);
    (void)hipMemsetAsync(bnst, 0, sizeof(float) * 4 * DD, stream);

    init_minmax_kernel<<<nb((long long)NN * MSGD, BS), BS, 0, stream>>>(
        maxb, minb, NN * MSGD);
    pe_embed_kernel<<<nb((long long)NN * DD, BS), BS, 0, stream>>>(
        x_cur, PE, pe_w + (size_t)l * PED * DD, pe_b + l * DD, xp);
    edge_agg_kernel<<<nb((long long)EE * MSGD, BS), BS, 0, stream>>>(
        ei, eattr, xp, sumb, sqb, maxb, minb, cntb);
    finalize_agg_kernel<<<nb((long long)NN * MSGD, BS), BS, 0, stream>>>(
        sumb, sqb, maxb, minb, cntb, aggh, s1, s2);
    prep_w_kernel<<<nb((long long)SCLD * DD, BS), BS, 0, stream>>>(
        conv_w + (size_t)l * SCLD * DD, wt);
    pna_gemm_wmma<<<NN / 16, 160, 0, stream>>>(
        aggh, wt, s1, s2, conv_b + l * DD, outbuf);
    bn_stats_kernel<<<nb((long long)NN * DD, BS), BS, 0, stream>>>(
        outbuf, bnst, NN * DD);
    bn_finalize_kernel<<<1, 96, 0, stream>>>(
        bnst, bn_g + l * DD, bn_b + l * DD, 1.0f / (float)NN);
    bn_apply_kernel<<<nb((long long)NN * DD, BS), BS, 0, stream>>>(
        outbuf, bnst, x_cur, NN * DD);
  }

  (void)hipMemsetAsync(pooled, 0, sizeof(float) * GG * DD, stream);
  (void)hipMemsetAsync(gcnt,   0, sizeof(float) * GG, stream);
  pool_kernel<<<nb((long long)NN * DD, BS), BS, 0, stream>>>(
      x_cur, batch, pooled, gcnt, NN * DD);
  head_kernel<<<GG / 64, 64, 0, stream>>>(
      pooled, gcnt, w1, b1, w2, b2, w3, b3, outp);
}
